// Seq2SeqModel_39539468927128
// MI455X (gfx1250) — matrix operand order repaired and checked by reading, not compiled, not run on hardware
//
#include <hip/hip_runtime.h>

// ---------------------------------------------------------------------------
// Seq2Seq GRU (B=128, RNN=2048, 3R=6144, IO=128, 49 enc steps + 25 dec steps)
// CDNA5 / gfx1250: wave32, v_wmma_f32_16x16x32_bf16, fp32 accumulate.
// Software-pipelined K loop: next iteration's fragments are loaded before the
// current iteration's WMMAs so s_wait_loadcnt never has to drain to zero.
// ---------------------------------------------------------------------------

typedef __attribute__((ext_vector_type(16))) __bf16 v16bf;
typedef __attribute__((ext_vector_type(8)))  __bf16 v8bf;
typedef __attribute__((ext_vector_type(8)))  float  v8f;

#define BATCH   128
#define RNN     2048
#define G3      6144      // 3*RNN
#define IOSZ    128       // IN_SIZE == OUT_SIZE
#define INSEQ   50
#define TGTSEQ  25
#define PFD     256       // prefetch distance in bf16 elements (512 B)

static __device__ __forceinline__ float sigmoidf_(float x) {
    return 1.0f / (1.0f + __expf(-x));
}

static __device__ __forceinline__ v8f wmma_bf16(v16bf a, v16bf b, v8f c) {
    // D = A(16x32 bf16) * B(32x16 bf16) + C(16x16 f32)
    return __builtin_amdgcn_wmma_f32_16x16x32_bf16(
        /*neg_a=*/false, a, /*neg_b=*/false, b,
        /*c_mod=*/(short)0, c, /*reuse_a=*/false, /*reuse_b=*/false);
}

// A fragment (16x32, bf16): lane l<16 -> row M=l, K=0..15;
// lane l>=16 -> row M=l-16, K=16..31.  Two 16B loads.
static __device__ __forceinline__ v16bf load_a_frag(const __bf16* rowp, int k0, int koff) {
    v8bf a0 = *(const v8bf*)(rowp + k0 + 2 * koff);
    v8bf a1 = *(const v8bf*)(rowp + k0 + 2 * koff + 8);
    v16bf a;
#pragma unroll
    for (int i = 0; i < 8; ++i) { a[i] = a0[i]; a[i + 8] = a1[i]; }
    return a;
}

// B fragment (32x16, bf16), B[k][n] = W[n][k] (W row-major NxK):
// lane l<16 -> col N=l, K=0..15; lane l>=16 -> col N=l-16, K=16..31.
// 16 contiguous bf16 per lane -> one 32B (aligned) load pair.
static __device__ __forceinline__ v16bf load_b_frag(const __bf16* wrowp, int k0, int kb16) {
    return *(const v16bf*)(wrowp + k0 + kb16);
}

// --------------------------- setup kernels ---------------------------------

__global__ void k_f32_to_bf16(const float* __restrict__ s, __bf16* __restrict__ d, int n) {
    int i = blockIdx.x * blockDim.x + threadIdx.x;
    if (i < n) d[i] = (__bf16)s[i];
}

__global__ void k_zero_h(float* __restrict__ hf, __bf16* __restrict__ hbf) {
    int i = blockIdx.x * blockDim.x + threadIdx.x;  // BATCH*RNN threads
    hf[i]  = 0.0f;
    hbf[i] = (__bf16)0.0f;
}

// enc_tm[t][b][d] = encoder_inputs[b][t][:] . fc1_w[d][:] + fc1_b[d]
__global__ void k_fc1_enc(const float* __restrict__ ein,   // [B][INSEQ][IOSZ]
                          const float* __restrict__ w1,    // [IOSZ][IOSZ]
                          const float* __restrict__ b1,    // [IOSZ]
                          float*  __restrict__ enc_tm,     // [INSEQ][B][IOSZ]
                          __bf16* __restrict__ encbf_tm) {
    int i = blockIdx.x * blockDim.x + threadIdx.x;  // INSEQ*B*IOSZ
    int t = i >> 14;            // / (128*128)
    int b = (i >> 7) & 127;
    int d = i & 127;
    const float* x = ein + ((size_t)b * INSEQ + t) * IOSZ;
    const float* w = w1 + (size_t)d * IOSZ;
    float acc = b1[d];
#pragma unroll 8
    for (int k = 0; k < IOSZ; ++k) acc = fmaf(x[k], w[k], acc);
    enc_tm[i]   = acc;
    encbf_tm[i] = (__bf16)acc;
}

// --------------------------- fused GRU step --------------------------------
// One launch per timestep.  Grid: 128 blocks (jtile over RNN/16), 256 thr
// (8 waves = 8 batch tiles).  Each wave: 4 WMMA accumulators (r, z, i_n, h_n),
// pipelined K-loop over RNN for h@w_hh^T, K-loop over IOSZ for x@w_ih^T, then
// the GRU nonlinearity and h_new store (fp32 + bf16) in-register.
__global__ void __launch_bounds__(256)
k_gru_step(const __bf16* __restrict__ xbf,     // [B][IOSZ]
           const __bf16* __restrict__ hbf_in,  // [B][RNN]
           const float*  __restrict__ hf_in,   // [B][RNN]
           const __bf16* __restrict__ whh,     // [G3][RNN]
           const __bf16* __restrict__ wih,     // [G3][IOSZ]
           const float*  __restrict__ bih,     // [G3]
           const float*  __restrict__ bhh,     // [G3]
           float*  __restrict__ hf_out,        // [B][RNN]
           __bf16* __restrict__ hbf_out) {
    const int lane  = threadIdx.x & 31;
    const int mtile = threadIdx.x >> 5;   // 0..7 (batch tiles)
    const int jtile = blockIdx.x;         // 0..127 (hidden tiles)

    const int lhalf = lane & 15;
    const int lhi   = lane >> 4;
    const int koff  = lhi * 8;            // A-fragment K offset
    const int kb16  = lhi * 16;           // B-fragment K offset

    const int arow = mtile * 16 + lhalf;          // batch row for A loads
    const int jrow = jtile * 16 + lhalf;          // hidden col for B loads

    const __bf16* hrow = hbf_in + (size_t)arow * RNN;
    const __bf16* br_r = whh + (size_t)(jrow)          * RNN;
    const __bf16* br_z = whh + (size_t)(jrow + RNN)    * RNN;
    const __bf16* br_n = whh + (size_t)(jrow + 2*RNN)  * RNN;

    v8f acc_r = {}, acc_z = {}, acc_in = {}, acc_hn = {};

    // ---- gh = h @ w_hh^T   (K = 2048), software-pipelined -----------------
    v16bf a  = load_a_frag(hrow, 0, koff);
    v16bf b0 = load_b_frag(br_r, 0, kb16);
    v16bf b1 = load_b_frag(br_z, 0, kb16);
    v16bf b2 = load_b_frag(br_n, 0, kb16);

#pragma unroll 2
    for (int k0 = 0; k0 < RNN; k0 += 32) {
        const int kn = (k0 + 32) & (RNN - 1);   // wraps to 0 on last iter
        // L0 prefetch a few chunks ahead (weights are L2-resident).
        __builtin_prefetch(br_r + ((k0 + PFD) & (RNN - 1)), 0, 3);
        __builtin_prefetch(br_z + ((k0 + PFD) & (RNN - 1)), 0, 3);
        __builtin_prefetch(br_n + ((k0 + PFD) & (RNN - 1)), 0, 3);
        // Issue next iteration's fragment loads before current WMMAs.
        v16bf an  = load_a_frag(hrow, kn, koff);
        v16bf b0n = load_b_frag(br_r, kn, kb16);
        v16bf b1n = load_b_frag(br_z, kn, kb16);
        v16bf b2n = load_b_frag(br_n, kn, kb16);
        acc_r  = wmma_bf16(a, b0, acc_r);
        acc_z  = wmma_bf16(a, b1, acc_z);
        acc_hn = wmma_bf16(a, b2, acc_hn);
        a = an; b0 = b0n; b1 = b1n; b2 = b2n;
    }

    // ---- gi = x @ w_ih^T   (K = 128) --------------------------------------
    const __bf16* xrow = xbf + (size_t)arow * IOSZ;
    const __bf16* bi_r = wih + (size_t)(jrow)         * IOSZ;
    const __bf16* bi_z = wih + (size_t)(jrow + RNN)   * IOSZ;
    const __bf16* bi_n = wih + (size_t)(jrow + 2*RNN) * IOSZ;
#pragma unroll
    for (int k0 = 0; k0 < IOSZ; k0 += 32) {
        v16bf ax  = load_a_frag(xrow, k0, koff);
        v16bf c0 = load_b_frag(bi_r, k0, kb16);
        v16bf c1 = load_b_frag(bi_z, k0, kb16);
        v16bf c2 = load_b_frag(bi_n, k0, kb16);
        acc_r  = wmma_bf16(ax, c0, acc_r);   // i_r + h_r share accumulator
        acc_z  = wmma_bf16(ax, c1, acc_z);   // i_z + h_z share accumulator
        acc_in = wmma_bf16(ax, c2, acc_in);  // i_n kept separate from h_n
    }

    // C/D layout: lane l<16 -> N=l, M=mbase+r(0..7); lane>=16 -> N=l-16, M=mbase+8+r
    const int ncol = jtile * 16 + lhalf;          // hidden index j
    const int m0   = mtile * 16 + lhi * 8;        // batch base row
    const float b_r  = bih[ncol]           + bhh[ncol];
    const float b_z  = bih[ncol + RNN]     + bhh[ncol + RNN];
    const float b_in = bih[ncol + 2*RNN];
    const float b_hn = bhh[ncol + 2*RNN];

#pragma unroll
    for (int r = 0; r < 8; ++r) {
        const int m = m0 + r;
        float rg = sigmoidf_(acc_r[r]  + b_r);
        float zg = sigmoidf_(acc_z[r]  + b_z);
        float ng = tanhf(acc_in[r] + b_in + rg * (acc_hn[r] + b_hn));
        float ho = hf_in[(size_t)m * RNN + ncol];
        float hn = (1.0f - zg) * ng + zg * ho;
        hf_out [(size_t)m * RNN + ncol] = hn;
        hbf_out[(size_t)m * RNN + ncol] = (__bf16)hn;
    }
}

// ------------------------- decoder fc2 + residual --------------------------
// out[b][k] = inp[b][k] + h[b][:] . fc2_w[k][:] + fc2_b[k]
// Writes out slice of d_out, and the next-step inp (fp32 + bf16).
__global__ void __launch_bounds__(256)
k_fc2_residual(const __bf16* __restrict__ hbf,     // [B][RNN]
               const __bf16* __restrict__ w2,      // [IOSZ][RNN]
               const float*  __restrict__ b2,      // [IOSZ]
               const float*  __restrict__ inp_in,  // [B][IOSZ]
               float*  __restrict__ out,           // d_out + t*IOSZ, row stride TGTSEQ*IOSZ
               float*  __restrict__ inp_out,       // [B][IOSZ]
               __bf16* __restrict__ inpbf_out) {
    const int lane = threadIdx.x & 31;
    const int gw   = blockIdx.x * 8 + (threadIdx.x >> 5);  // 64 waves
    const int mtile = gw >> 3;   // batch tile 0..7
    const int ntile = gw & 7;    // output-col tile 0..7

    const int lhalf = lane & 15;
    const int lhi   = lane >> 4;
    const int koff  = lhi * 8;
    const int kb16  = lhi * 16;

    const __bf16* hrow = hbf + (size_t)(mtile * 16 + lhalf) * RNN;
    const __bf16* wrow = w2  + (size_t)(ntile * 16 + lhalf) * RNN;

    v8f acc = {};
    v16bf a = load_a_frag(hrow, 0, koff);
    v16bf b = load_b_frag(wrow, 0, kb16);
#pragma unroll 2
    for (int k0 = 0; k0 < RNN; k0 += 32) {
        const int kn = (k0 + 32) & (RNN - 1);
        __builtin_prefetch(wrow + ((k0 + PFD) & (RNN - 1)), 0, 3);
        v16bf an = load_a_frag(hrow, kn, koff);
        v16bf bn = load_b_frag(wrow, kn, kb16);
        acc = wmma_bf16(a, b, acc);
        a = an; b = bn;
    }

    const int ncol = ntile * 16 + lhalf;
    const int m0   = mtile * 16 + lhi * 8;
    const float bb = b2[ncol];
#pragma unroll
    for (int r = 0; r < 8; ++r) {
        const int m = m0 + r;
        float v = acc[r] + bb + inp_in[(size_t)m * IOSZ + ncol];
        out[(size_t)m * (TGTSEQ * IOSZ) + ncol] = v;
        inp_out  [(size_t)m * IOSZ + ncol] = v;
        inpbf_out[(size_t)m * IOSZ + ncol] = (__bf16)v;
    }
}

// ------------------------------- host --------------------------------------

extern "C" void kernel_launch(void* const* d_in, const int* in_sizes, int n_in,
                              void* d_out, int out_size, void* d_ws, size_t ws_size,
                              hipStream_t stream) {
    const float* enc_in = (const float*)d_in[0];
    // d_in[1] decoder_inputs: unused by the reference forward
    const float* fc1_w = (const float*)d_in[2];
    const float* fc1_b = (const float*)d_in[3];
    const float* w_ih  = (const float*)d_in[4];
    const float* w_hh  = (const float*)d_in[5];
    const float* b_ih  = (const float*)d_in[6];
    const float* b_hh  = (const float*)d_in[7];
    const float* fc2_w = (const float*)d_in[8];
    const float* fc2_b = (const float*)d_in[9];
    float* out = (float*)d_out;

    char* ws = (char*)d_ws;
    size_t off = 0;
    auto alloc = [&](size_t bytes) -> char* {
        char* p = ws + off;
        off = (off + bytes + 255) & ~(size_t)255;
        return p;
    };

    __bf16* whh_bf   = (__bf16*)alloc((size_t)G3 * RNN * 2);        // 25.2 MB
    __bf16* wih_bf   = (__bf16*)alloc((size_t)G3 * IOSZ * 2);       //  1.6 MB
    __bf16* w2_bf    = (__bf16*)alloc((size_t)IOSZ * RNN * 2);      //  0.5 MB
    float*  enc_tm   = (float*) alloc((size_t)INSEQ * BATCH * IOSZ * 4);
    __bf16* encbf_tm = (__bf16*)alloc((size_t)INSEQ * BATCH * IOSZ * 2);
    float*  hf[2]    = { (float*) alloc((size_t)BATCH * RNN * 4),
                         (float*) alloc((size_t)BATCH * RNN * 4) };
    __bf16* hbf[2]   = { (__bf16*)alloc((size_t)BATCH * RNN * 2),
                         (__bf16*)alloc((size_t)BATCH * RNN * 2) };
    float*  inp_f    = (float*) alloc((size_t)BATCH * IOSZ * 4);
    __bf16* inp_bf   = (__bf16*)alloc((size_t)BATCH * IOSZ * 2);
    (void)ws_size; (void)in_sizes; (void)n_in; (void)out_size;

    // Weight conversions (L2-resident thereafter: 27 MB << 192 MB L2).
    {
        int n = G3 * RNN;
        k_f32_to_bf16<<<(n + 255) / 256, 256, 0, stream>>>(w_hh, whh_bf, n);
        n = G3 * IOSZ;
        k_f32_to_bf16<<<(n + 255) / 256, 256, 0, stream>>>(w_ih, wih_bf, n);
        n = IOSZ * RNN;
        k_f32_to_bf16<<<(n + 255) / 256, 256, 0, stream>>>(fc2_w, w2_bf, n);
    }

    // fc1 + time-major transpose (+ bf16 copy); h0 = 0.
    k_fc1_enc<<<(INSEQ * BATCH * IOSZ) / 256, 256, 0, stream>>>(
        enc_in, fc1_w, fc1_b, enc_tm, encbf_tm);
    k_zero_h<<<(BATCH * RNN) / 256, 256, 0, stream>>>(hf[0], hbf[0]);

    int cur = 0;

    // Encoder: 49 GRU steps over enc_t[0..48].
    for (int t = 0; t < INSEQ - 1; ++t) {
        k_gru_step<<<RNN / 16, 256, 0, stream>>>(
            encbf_tm + (size_t)t * BATCH * IOSZ,
            hbf[cur], hf[cur], whh_bf, wih_bf, b_ih, b_hh,
            hf[1 - cur], hbf[1 - cur]);
        cur ^= 1;
    }

    // Decoder: autoregressive, inp0 = enc_t[-1].
    const __bf16* x_bf   = encbf_tm + (size_t)(INSEQ - 1) * BATCH * IOSZ;
    const float*  x_f32  = enc_tm   + (size_t)(INSEQ - 1) * BATCH * IOSZ;
    for (int t = 0; t < TGTSEQ; ++t) {
        k_gru_step<<<RNN / 16, 256, 0, stream>>>(
            x_bf, hbf[cur], hf[cur], whh_bf, wih_bf, b_ih, b_hh,
            hf[1 - cur], hbf[1 - cur]);
        cur ^= 1;
        k_fc2_residual<<<8, 256, 0, stream>>>(
            hbf[cur], w2_bf, fc2_b, x_f32,
            out + (size_t)t * IOSZ, inp_f, inp_bf);
        x_bf  = inp_bf;
        x_f32 = inp_f;
    }
}
